// BasicModule_46540265620048
// MI455X (gfx1250) — matrix operand order repaired and checked by reading, not compiled, not run on hardware
//
#include <hip/hip_runtime.h>
#include <hip/hip_bf16.h>

#define DEPTH 20
#define FEAT  2048
#define HID   1024
#define HALF  1024
#define NB    4096
#define CLAMPV 2.0f
#define TWO_OVER_PI 0.6366197723675814f

typedef __attribute__((ext_vector_type(16))) __bf16 bf16x16;
typedef __attribute__((ext_vector_type(8)))  float  f32x8;

#define LDS_STRIDE 40   // 32 K elems + 8 pad (80B rows -> conflict-free 16B frag loads)

// ---------------------------------------------------------------------------
// fragment loaders from LDS (row-major [row][k], stride LDS_STRIDE)
// A operand: lane<16 holds K {0..7,16..23}; lane>=16 holds K {8..15,24..31}
// B operand: lane<16 holds K {0..7,16..23}; lane>=16 holds K {8..15,24..31}
// ---------------------------------------------------------------------------
__device__ __forceinline__ bf16x16 load_fragA(const __bf16* rowp, int lane) {
    const int kb = (lane >> 4) << 3;              // 0 or 8
    union { uint4 u[2]; bf16x16 v; } f;
    f.u[0] = *reinterpret_cast<const uint4*>(rowp + kb);
    f.u[1] = *reinterpret_cast<const uint4*>(rowp + kb + 16);
    return f.v;
}
__device__ __forceinline__ bf16x16 load_fragB(const __bf16* rowp, int lane) {
    const int kb = (lane >> 4) << 3;              // 0 or 8
    union { uint4 u[2]; bf16x16 v; } f;
    f.u[0] = *reinterpret_cast<const uint4*>(rowp + kb);
    f.u[1] = *reinterpret_cast<const uint4*>(rowp + kb + 16);
    return f.v;
}

// ---------------------------------------------------------------------------
// GEMM: C[M x N] = A[M x K](bf16) * B[K x N](fp32, converted to bf16 on load)
//        + bias[N]; EPI==0 -> leaky-relu, store bf16 ; EPI==1 -> store fp32
// Block: 256 thr (8 waves), tile 128x128, BK=32, double-buffered LDS.
// ---------------------------------------------------------------------------
template<int EPI>
__global__ __launch_bounds__(256) void gemm_bf16_k(
    const unsigned short* __restrict__ A_, const float* __restrict__ B,
    const float* __restrict__ bias,
    unsigned short* __restrict__ outB_, float* __restrict__ outF,
    int N, int K)
{
    const __bf16* A   = reinterpret_cast<const __bf16*>(A_);
    __bf16* outB      = reinterpret_cast<__bf16*>(outB_);

    __shared__ __align__(16) __bf16 sA[2][128][LDS_STRIDE];
    __shared__ __align__(16) __bf16 sB[2][128][LDS_STRIDE];

    const int tid  = threadIdx.x;
    const int lane = tid & 31;
    const int w    = tid >> 5;
    const int wm   = w & 3;              // wave grid 4 (M) x 2 (N)
    const int wn   = w >> 2;
    const int lr   = lane & 15;

    const int blockM = blockIdx.y * 128;
    const int blockN = blockIdx.x * 128;

    // A staging: thread -> (row 0..127, k-half 0/16), 2x 16B loads
    const int arow = tid >> 1;
    const int akh  = (tid & 1) << 4;
    // B staging: thread -> (col 0..127, k-group 0/16), 16 strided dwords
    const int bn   = tid & 127;
    const int bkg  = (tid >> 7) << 4;

    const __bf16* Aptr = A + (size_t)(blockM + arow) * K + akh;
    const float*  Bptr = B + (size_t)bkg * N + blockN + bn;

    f32x8 acc[2][4] = {};

    uint4 a0, a1; float br[16];
    // prologue: fill buffer 0
    a0 = *reinterpret_cast<const uint4*>(Aptr);
    a1 = *reinterpret_cast<const uint4*>(Aptr + 8);
    #pragma unroll
    for (int kk = 0; kk < 16; kk++) br[kk] = Bptr[(size_t)kk * N];
    {
        *reinterpret_cast<uint4*>(&sA[0][arow][akh])     = a0;
        *reinterpret_cast<uint4*>(&sA[0][arow][akh + 8]) = a1;
        union { __bf16 b[16]; uint4 u[2]; } pb;
        #pragma unroll
        for (int kk = 0; kk < 16; kk++) pb.b[kk] = (__bf16)br[kk];
        *reinterpret_cast<uint4*>(&sB[0][bn][bkg])     = pb.u[0];
        *reinterpret_cast<uint4*>(&sB[0][bn][bkg + 8]) = pb.u[1];
    }
    __syncthreads();

    for (int kt = 0; kt < K; kt += 32) {
        const int  cur  = (kt >> 5) & 1;
        const bool more = (kt + 32) < K;
        if (more) {   // issue next-tile global loads (overlap with WMMA below)
            const __bf16* Ap = Aptr + kt + 32;
            a0 = *reinterpret_cast<const uint4*>(Ap);
            a1 = *reinterpret_cast<const uint4*>(Ap + 8);
            const float* Bp = Bptr + (size_t)(kt + 32) * N;
            #pragma unroll
            for (int kk = 0; kk < 16; kk++) br[kk] = Bp[(size_t)kk * N];
        }
        bf16x16 af[2], bf[4];
        #pragma unroll
        for (int i = 0; i < 2; i++)
            af[i] = load_fragA(&sA[cur][wm * 32 + i * 16 + lr][0], lane);
        #pragma unroll
        for (int j = 0; j < 4; j++)
            bf[j] = load_fragB(&sB[cur][wn * 64 + j * 16 + lr][0], lane);
        #pragma unroll
        for (int i = 0; i < 2; i++)
            #pragma unroll
            for (int j = 0; j < 4; j++)
                acc[i][j] = __builtin_amdgcn_wmma_f32_16x16x32_bf16(
                    false, af[i], false, bf[j], (short)0, acc[i][j], false, false);
        if (more) {   // commit staged tile into the other LDS buffer
            const int nxt = cur ^ 1;
            *reinterpret_cast<uint4*>(&sA[nxt][arow][akh])     = a0;
            *reinterpret_cast<uint4*>(&sA[nxt][arow][akh + 8]) = a1;
            union { __bf16 b[16]; uint4 u[2]; } pb;
            #pragma unroll
            for (int kk = 0; kk < 16; kk++) pb.b[kk] = (__bf16)br[kk];
            *reinterpret_cast<uint4*>(&sB[nxt][bn][bkg])     = pb.u[0];
            *reinterpret_cast<uint4*>(&sB[nxt][bn][bkg + 8]) = pb.u[1];
        }
        __syncthreads();
    }

    // epilogue: C layout -> n = lane%16, m = vgpr + 8*(lane/16)
    const int mh = (lane >> 4) << 3;
    #pragma unroll
    for (int i = 0; i < 2; i++) {
        const int m0 = blockM + wm * 32 + i * 16 + mh;
        #pragma unroll
        for (int j = 0; j < 4; j++) {
            const int n  = blockN + wn * 64 + j * 16 + lr;
            const float bv = bias[n];
            #pragma unroll
            for (int v = 0; v < 8; v++) {
                float val = acc[i][j][v] + bv;
                const size_t off = (size_t)(m0 + v) * N + n;
                if (EPI == 0) {
                    val = (val >= 0.f) ? val : 0.01f * val;   // leaky relu
                    outB[off] = (__bf16)val;
                } else {
                    outF[off] = val;
                }
            }
        }
    }
}

// ---------------------------------------------------------------------------
// permute: zp[m][j] = z[m][perm[j]]; also emit bf16(x2) for the first GEMM
// ---------------------------------------------------------------------------
__global__ __launch_bounds__(256) void permute_k(
    const float* __restrict__ z, const int* __restrict__ perm,
    float* __restrict__ zp, unsigned short* __restrict__ xb_)
{
    __bf16* xb = reinterpret_cast<__bf16*>(xb_);
    const int m = blockIdx.y;
    const int j = blockIdx.x * 256 + threadIdx.x;
    const float v = z[(size_t)m * FEAT + perm[j]];
    zp[(size_t)m * FEAT + j] = v;
    if (j >= HALF) xb[(size_t)m * HALF + (j - HALF)] = (__bf16)v;
}

// ---------------------------------------------------------------------------
// coupling: e = C*(2/pi)*atan(r[:, :H]/C); y = exp(e)*xo + r[:, H:];
//           jac[m] += sum_j e.  One block per row (no atomics needed).
// ---------------------------------------------------------------------------
__global__ __launch_bounds__(256) void coupling_k(
    const float* __restrict__ r, const float* __restrict__ xo,
    float* __restrict__ yf, unsigned short* __restrict__ yb_,
    float* __restrict__ jac)
{
    __bf16* yb = reinterpret_cast<__bf16*>(yb_);
    const int m = blockIdx.x;
    const int t = threadIdx.x;
    float s = 0.f;
    #pragma unroll
    for (int c = 0; c < 4; c++) {
        const int j = t + c * 256;
        const float rv = r[(size_t)m * FEAT + j];
        const float e  = CLAMPV * TWO_OVER_PI * atanf(rv * (1.0f / CLAMPV));
        const float y  = expf(e) * xo[(size_t)m * FEAT + j] + r[(size_t)m * FEAT + HALF + j];
        yf[(size_t)m * FEAT + j] = y;
        if (yb) yb[(size_t)m * HALF + j] = (__bf16)y;
        s += e;
    }
    __shared__ float red[256];
    red[t] = s; __syncthreads();
    #pragma unroll
    for (int off = 128; off > 0; off >>= 1) {
        if (t < off) red[t] += red[t + off];
        __syncthreads();
    }
    if (t == 0) jac[m] += red[0];
}

__global__ void zero_jac_k(float* jac) { jac[blockIdx.x * 256 + threadIdx.x] = 0.f; }

// ---------------------------------------------------------------------------
extern "C" void kernel_launch(void* const* d_in, const int* in_sizes, int n_in,
                              void* d_out, int out_size, void* d_ws, size_t ws_size,
                              hipStream_t stream) {
    (void)in_sizes; (void)n_in; (void)out_size; (void)ws_size;
    const float* x     = (const float*)d_in[0];
    const int*   perms = (const int*)  d_in[1];
    const float* s1_w1 = (const float*)d_in[2];
    const float* s1_b1 = (const float*)d_in[3];
    const float* s1_w2 = (const float*)d_in[4];
    const float* s1_b2 = (const float*)d_in[5];
    const float* s2_w1 = (const float*)d_in[6];
    const float* s2_b1 = (const float*)d_in[7];
    const float* s2_w2 = (const float*)d_in[8];
    const float* s2_b2 = (const float*)d_in[9];

    float* z   = (float*)d_out;                  // master state lives in d_out
    float* jac = z + (size_t)NB * FEAT;

    char* ws = (char*)d_ws;
    float* zp = (float*)ws;                      ws += (size_t)NB * FEAT * sizeof(float);
    float* r  = (float*)ws;                      ws += (size_t)NB * FEAT * sizeof(float);
    unsigned short* xb = (unsigned short*)ws;    ws += (size_t)NB * HALF * sizeof(unsigned short);
    unsigned short* h  = (unsigned short*)ws;

    hipMemcpyAsync(z, x, (size_t)NB * FEAT * sizeof(float),
                   hipMemcpyDeviceToDevice, stream);
    zero_jac_k<<<NB / 256, 256, 0, stream>>>(jac);

    const dim3 g1(HID  / 128, NB / 128);   // 8 x 32 blocks
    const dim3 g2(FEAT / 128, NB / 128);   // 16 x 32 blocks

    for (int d = 0; d < DEPTH; d++) {
        permute_k<<<dim3(FEAT / 256, NB), 256, 0, stream>>>(
            z, perms + (size_t)d * FEAT, zp, xb);                 // xb = bf16(x2)
        // r2 = subnet(x2; s2_*)
        gemm_bf16_k<0><<<g1, 256, 0, stream>>>(
            xb, s2_w1 + (size_t)d * HALF * HID, s2_b1 + (size_t)d * HID,
            h, nullptr, HID, HALF);
        gemm_bf16_k<1><<<g2, 256, 0, stream>>>(
            h, s2_w2 + (size_t)d * HID * FEAT, s2_b2 + (size_t)d * FEAT,
            nullptr, r, FEAT, HID);
        // y1 = exp(e2)*x1 + r2[:,H:] ; jac += sum(e2); xb = bf16(y1)
        coupling_k<<<NB, 256, 0, stream>>>(r, zp, z, xb, jac);
        // r1 = subnet(y1; s1_*)
        gemm_bf16_k<0><<<g1, 256, 0, stream>>>(
            xb, s1_w1 + (size_t)d * HALF * HID, s1_b1 + (size_t)d * HID,
            h, nullptr, HID, HALF);
        gemm_bf16_k<1><<<g2, 256, 0, stream>>>(
            h, s1_w2 + (size_t)d * HID * FEAT, s1_b2 + (size_t)d * FEAT,
            nullptr, r, FEAT, HID);
        // y2 = exp(e1)*x2 + r1[:,H:] ; jac += sum(e1)
        coupling_k<<<NB, 256, 0, stream>>>(r, zp + HALF, z + HALF, nullptr, jac);
    }
}
